// TransMIL_27582279975349
// MI455X (gfx1250) — matrix operand order repaired and checked by reading, not compiled, not run on hardware
//
#include <hip/hip_runtime.h>
#include <hip/hip_bf16.h>
#include <math.h>

typedef _Float16 half_t;
typedef __attribute__((ext_vector_type(16))) _Float16 v16h;
typedef __attribute__((ext_vector_type(8)))  _Float16 v8h;
typedef __attribute__((ext_vector_type(8)))  float    v8f;

// ---------------- problem constants ----------------
constexpr int kB   = 8;
constexpr int kBAG = 4096;
constexpr int kIN  = 1024;
constexpr int kEMB = 512;
constexpr int kH   = 4;
constexpr int kDH  = 128;
constexpr int kSEQ = 4352;   // 4097 padded to multiple of 256
constexpr int kPAD = 255;    // front pad rows
constexpr int kNL  = 256;    // landmarks (M)
constexpr int kL   = 17;     // seq / landmarks
constexpr int kBH  = 32;     // B * HEADS
constexpr int kQKV = 1536;
constexpr int kPS  = 64;     // 64x64 PPEG grid
constexpr int kRES = 33;     // residual conv taps

// ---------------- WMMA GEMM ----------------
// C[M,N] (f32) = A[M,K](f16) @ B(f16) (+bias), batched via grid.z.
// TRANSB=0: B is [K,N] row-major (ldb=N).  TRANSB=1: B is [N,K] row-major (ldb=K).
// ALL shapes in this network are tile-exact: M%128==0, N%128==0, K%64==0 -> no guards.
#define BMT 128
#define BNT 128
#define BKT 64
#define LDS_S 72   // padded row stride (halves); 144B keeps 16B alignment

__device__ inline v16h ld16x2(const half_t* lo, const half_t* hi) {
  v8h a = *(const v8h*)lo;
  v8h b = *(const v8h*)hi;
  v16h r;
#pragma unroll
  for (int i = 0; i < 8; ++i) { r[i] = a[i]; r[i + 8] = b[i]; }
  return r;
}

template <bool TRANSB, bool HAS_BIAS>
__global__ __launch_bounds__(256) void wmma_gemm(
    const half_t* __restrict__ A, const half_t* __restrict__ Bm,
    const float* __restrict__ bias, float* __restrict__ C,
    int M, int N, int K, int lda, int ldb, int ldc,
    long long sA, long long sB, long long sC) {
  __shared__ __align__(16) half_t As[BMT * LDS_S];
  __shared__ __align__(16) half_t Bs[BNT * LDS_S];

  const int tid  = threadIdx.x;
  const int lane = tid & 31;
  const int wid  = tid >> 5;
  const int wm   = wid & 3;   // 4 waves along M (32 rows each)
  const int wn   = wid >> 2;  // 2 waves along N (64 cols each)
  const int hsel = lane >> 4;
  const int lmod = lane & 15;

  const long long bz = blockIdx.z;
  const half_t* Ab = A  + bz * sA;
  const half_t* Bb = Bm + bz * sB;
  float*        Cb = C  + bz * sC;

  const int m0 = blockIdx.y * BMT;
  const int n0 = blockIdx.x * BNT;

  v8f zero = {};
  v8f acc[2][4];
#pragma unroll
  for (int i = 0; i < 2; ++i)
#pragma unroll
    for (int j = 0; j < 4; ++j) acc[i][j] = zero;

  for (int k0 = 0; k0 < K; k0 += BKT) {
    // ---- stage A tile (128x64 halves): load 4 chunks, then store 4 chunks ----
    {
      v8h av[4];
#pragma unroll
      for (int it = 0; it < 4; ++it) {
        int c    = tid * 4 + it;     // 0..1023 chunks of 8 halves
        int row  = c >> 3;           // 0..127
        int col8 = (c & 7) << 3;     // 0..56
        av[it] = *(const v8h*)(Ab + (long long)(m0 + row) * lda + k0 + col8);
      }
#pragma unroll
      for (int it = 0; it < 4; ++it) {
        int c    = tid * 4 + it;
        int row  = c >> 3;
        int col8 = (c & 7) << 3;
        *(v8h*)(As + row * LDS_S + col8) = av[it];
      }
    }
    // ---- stage B tile as Bs[n][k]: load phase then store phase ----
    if (TRANSB) {
      v8h bv[4];
#pragma unroll
      for (int it = 0; it < 4; ++it) {
        int c    = tid * 4 + it;
        int row  = c >> 3;            // n: 0..127
        int col8 = (c & 7) << 3;      // k: 0..56
        bv[it] = *(const v8h*)(Bb + (long long)(n0 + row) * ldb + k0 + col8);
      }
#pragma unroll
      for (int it = 0; it < 4; ++it) {
        int c    = tid * 4 + it;
        int row  = c >> 3;
        int col8 = (c & 7) << 3;
        *(v8h*)(Bs + row * LDS_S + col8) = bv[it];
      }
    } else {
      v8h bv[4];
#pragma unroll
      for (int it = 0; it < 4; ++it) {
        int c  = tid * 4 + it;        // 0..1023
        int k  = c >> 4;              // 0..63
        int n8 = (c & 15) << 3;       // 0..120
        bv[it] = *(const v8h*)(Bb + (long long)(k0 + k) * ldb + n0 + n8);
      }
#pragma unroll
      for (int it = 0; it < 4; ++it) {
        int c  = tid * 4 + it;
        int k  = c >> 4;
        int n8 = (c & 15) << 3;
#pragma unroll
        for (int q = 0; q < 8; ++q) Bs[(n8 + q) * LDS_S + k] = bv[it][q];
      }
    }
    if (k0 + BKT < K)  // hint next A tile toward the caches
      __builtin_prefetch(Ab + (long long)(m0 + (tid >> 1)) * lda + k0 + BKT, 0, 1);
    __syncthreads();

    // ---- two K=32 sub-steps per stage: 16 WMMAs between barriers ----
#pragma unroll
    for (int ks = 0; ks < 2; ++ks) {
      const int kk = ks * 32;
      v16h afr[2], bfr[4];
#pragma unroll
      for (int mi = 0; mi < 2; ++mi) {
        const half_t* base = As + (wm * 32 + mi * 16 + lmod) * LDS_S + kk;
        afr[mi] = ld16x2(base + 8 * hsel, base + 16 + 8 * hsel);
      }
#pragma unroll
      for (int ni = 0; ni < 4; ++ni) {
        const half_t* base = Bs + (wn * 64 + ni * 16 + lmod) * LDS_S + kk;
        bfr[ni] = ld16x2(base + 8 * hsel, base + 16 + 8 * hsel);
      }
#pragma unroll
      for (int mi = 0; mi < 2; ++mi)
#pragma unroll
        for (int ni = 0; ni < 4; ++ni)
          acc[mi][ni] = __builtin_amdgcn_wmma_f32_16x16x32_f16(
              false, afr[mi], false, bfr[ni], (short)0, acc[mi][ni], false, false);
    }
    __syncthreads();
  }

  // ---- epilogue (C/D layout: m = r + 8*hsel, n = lane%16), unguarded ----
#pragma unroll
  for (int mi = 0; mi < 2; ++mi)
#pragma unroll
    for (int ni = 0; ni < 4; ++ni) {
      int col = n0 + wn * 64 + ni * 16 + lmod;
      float bv = HAS_BIAS ? bias[col] : 0.f;
#pragma unroll
      for (int r = 0; r < 8; ++r) {
        int row = m0 + wm * 32 + mi * 16 + r + 8 * hsel;
        Cb[(long long)row * ldc + col] = acc[mi][ni][r] + bv;
      }
    }
}

// ---------------- elementwise / support kernels ----------------
__global__ void cvt_scale_k(const float* __restrict__ X, half_t* __restrict__ O,
                            float s, long long n) {
  long long i = (long long)blockIdx.x * blockDim.x + threadIdx.x;
  long long st = (long long)gridDim.x * blockDim.x;
  for (; i < n; i += st) O[i] = (half_t)(X[i] * s);
}

__global__ __launch_bounds__(256) void layernorm_to_h_k(
    const float* __restrict__ S, const float* __restrict__ g,
    const float* __restrict__ b, half_t* __restrict__ O) {
  long long row = blockIdx.x;           // over kB*kSEQ
  int n = (int)(row % kSEQ);
  const float* x = S + row * kEMB;
  half_t* o = O + row * kEMB;
  int t = threadIdx.x;
  if (n < kPAD) {                       // padded rows -> exact zeros (matches jnp.pad)
    for (int c = t; c < kEMB; c += 256) o[c] = (half_t)0.f;
    return;
  }
  __shared__ float red[256];
  float s1 = 0.f, s2 = 0.f;
  for (int c = t; c < kEMB; c += 256) { float v = x[c]; s1 += v; s2 += v * v; }
  red[t] = s1; __syncthreads();
  for (int s = 128; s > 0; s >>= 1) { if (t < s) red[t] += red[t + s]; __syncthreads(); }
  float mu = red[0] * (1.f / kEMB); __syncthreads();
  red[t] = s2; __syncthreads();
  for (int s = 128; s > 0; s >>= 1) { if (t < s) red[t] += red[t + s]; __syncthreads(); }
  float var = red[0] * (1.f / kEMB) - mu * mu;
  float inv = rsqrtf(var + 1e-5f);
  for (int c = t; c < kEMB; c += 256)
    o[c] = (half_t)(((x[c] - mu) * inv) * g[c] + b[c]);
}

__global__ void split_heads_k(const float* __restrict__ qkv, half_t* __restrict__ q,
                              half_t* __restrict__ k, half_t* __restrict__ v, float scale) {
  long long i = (long long)blockIdx.x * blockDim.x + threadIdx.x;
  long long tot = (long long)kB * kSEQ * kEMB;
  if (i >= tot) return;
  int c = (int)(i % kEMB);
  long long bn = i / kEMB;
  int n = (int)(bn % kSEQ);
  int b = (int)(bn / kSEQ);
  int h = c >> 7, d = c & 127;
  long long src = ((long long)b * kSEQ + n) * kQKV + c;
  long long dst = (((long long)(b * kH + h)) * kSEQ + n) * kDH + d;
  q[dst] = (half_t)(qkv[src] * scale);
  k[dst] = (half_t)(qkv[src + kEMB]);
  v[dst] = (half_t)(qkv[src + 2 * kEMB]);
}

__global__ void landmark_mean_k(const half_t* __restrict__ x, half_t* __restrict__ o) {
  long long i = (long long)blockIdx.x * blockDim.x + threadIdx.x;  // kBH*kNL*kDH
  if (i >= (long long)kBH * kNL * kDH) return;
  int d = (int)(i & (kDH - 1));
  long long r = i >> 7;
  int m = (int)(r % kNL);
  int bh = (int)(r / kNL);
  float s = 0.f;
  for (int j = 0; j < kL; ++j)
    s += (float)x[(((long long)bh * kSEQ) + m * kL + j) * kDH + d];
  o[(((long long)bh * kNL) + m) * kDH + d] = (half_t)(s * (1.f / kL));
}

__global__ __launch_bounds__(256) void softmax_rows_k(float* __restrict__ X,
                                                      half_t* __restrict__ Oh, int cols) {
  long long row = blockIdx.x;
  float* x = X + row * cols;
  half_t* o = Oh + row * cols;
  __shared__ float red[256];
  int t = threadIdx.x;
  float mx = -1e30f;
  for (int c = t; c < cols; c += 256) mx = fmaxf(mx, x[c]);
  red[t] = mx; __syncthreads();
  for (int s = 128; s > 0; s >>= 1) { if (t < s) red[t] = fmaxf(red[t], red[t + s]); __syncthreads(); }
  mx = red[0]; __syncthreads();
  float sum = 0.f;
  for (int c = t; c < cols; c += 256) { float e = __expf(x[c] - mx); x[c] = e; sum += e; }
  red[t] = sum; __syncthreads();
  for (int s = 128; s > 0; s >>= 1) { if (t < s) red[t] += red[t + s]; __syncthreads(); }
  float inv = 1.f / red[0];
  for (int c = t; c < cols; c += 256) { float v = x[c] * inv; x[c] = v; o[c] = (half_t)v; }
}

__global__ __launch_bounds__(256) void pinv_init_k(const float* __restrict__ a2,
                                                   half_t* __restrict__ zh) {
  int b = blockIdx.x;                 // kBH batches of 256x256
  const float* X = a2 + (long long)b * kNL * kNL;
  half_t* Z = zh + (long long)b * kNL * kNL;
  __shared__ float red[256];
  int t = threadIdx.x;
  float rs = 0.f, cs = 0.f;
  for (int j = 0; j < kNL; ++j) { rs += X[t * kNL + j]; cs += X[j * kNL + t]; }
  red[t] = rs; __syncthreads();
  for (int s = 128; s > 0; s >>= 1) { if (t < s) red[t] = fmaxf(red[t], red[t + s]); __syncthreads(); }
  float m1 = red[0]; __syncthreads();
  red[t] = cs; __syncthreads();
  for (int s = 128; s > 0; s >>= 1) { if (t < s) red[t] = fmaxf(red[t], red[t + s]); __syncthreads(); }
  float inv = 1.f / (m1 * red[0]);
  for (int j = 0; j < kNL; ++j)       // z = X^T * inv
    Z[j * kNL + t] = (half_t)(X[t * kNL + j] * inv);
}

__global__ void aI_minus_k(const float* __restrict__ X, half_t* __restrict__ O,
                           float alpha, long long tot) {
  long long i = (long long)blockIdx.x * blockDim.x + threadIdx.x;
  if (i >= tot) return;
  int j = (int)(i % kNL);
  int r = (int)((i / kNL) % kNL);
  O[i] = (half_t)(((r == j) ? alpha : 0.f) - X[i]);
}

__global__ void merge_res_k(const float* __restrict__ attn, const half_t* __restrict__ v,
                            const float* __restrict__ rk, half_t* __restrict__ Mh) {
  long long i = (long long)blockIdx.x * blockDim.x + threadIdx.x;
  long long tot = (long long)kB * kSEQ * kEMB;
  if (i >= tot) return;
  int c = (int)(i % kEMB);
  long long bn = i / kEMB;
  int n = (int)(bn % kSEQ);
  int b = (int)(bn / kSEQ);
  int h = c >> 7, d = c & 127;
  long long base = ((long long)(b * kH + h)) * kSEQ * kDH + d;
  float s = attn[base + (long long)n * kDH];
  for (int t = 0; t < kRES; ++t) {
    int nn = n + t - kRES / 2;
    if (nn >= 0 && nn < kSEQ) s += rk[h * kRES + t] * (float)v[base + (long long)nn * kDH];
  }
  Mh[((long long)b * kSEQ + n) * kEMB + c] = (half_t)s;
}

__global__ void add_residual_k(float* __restrict__ S, const float* __restrict__ P) {
  long long i = (long long)blockIdx.x * blockDim.x + threadIdx.x;
  long long tot = (long long)kB * kSEQ * kEMB;
  if (i >= tot) return;
  int n = (int)((i / kEMB) % kSEQ);
  if (n >= kPAD) S[i] += P[i];        // out[:, -n:] residual
}

__global__ void set_cls_k(float* __restrict__ S, const float* __restrict__ cls) {
  int b = blockIdx.x;
  for (int c = threadIdx.x; c < kEMB; c += blockDim.x)
    S[((long long)b * kSEQ + kPAD) * kEMB + c] = cls[c];
}

__global__ void copy_cls_k(const float* __restrict__ Si, float* __restrict__ So) {
  int b = blockIdx.x;
  for (int c = threadIdx.x; c < kEMB; c += blockDim.x)
    So[((long long)b * kSEQ + kPAD) * kEMB + c] = Si[((long long)b * kSEQ + kPAD) * kEMB + c];
}

__global__ void ppeg_k(const float* __restrict__ Si, float* __restrict__ So,
                       const float* __restrict__ k7, const float* __restrict__ b7,
                       const float* __restrict__ k5, const float* __restrict__ b5,
                       const float* __restrict__ k3, const float* __restrict__ b3) {
  long long i = (long long)blockIdx.x * blockDim.x + threadIdx.x;
  long long tot = (long long)kB * kEMB * kPS * kPS;
  if (i >= tot) return;
  int xy = (int)(i % (kPS * kPS));
  int c  = (int)((i / (kPS * kPS)) % kEMB);
  int b  = (int)(i / ((long long)(kPS * kPS) * kEMB));
  int y = xy >> 6, x = xy & 63;
  auto F = [&](int yy, int xx) -> float {
    if (yy < 0 || yy >= kPS || xx < 0 || xx >= kPS) return 0.f;
    return Si[((long long)b * kSEQ + (kPAD + 1) + yy * kPS + xx) * kEMB + c];
  };
  float s = F(y, x) + b7[c] + b5[c] + b3[c];
  for (int dy = -3; dy <= 3; ++dy)
    for (int dx = -3; dx <= 3; ++dx)
      s += k7[c * 49 + (dy + 3) * 7 + (dx + 3)] * F(y + dy, x + dx);
  for (int dy = -2; dy <= 2; ++dy)
    for (int dx = -2; dx <= 2; ++dx)
      s += k5[c * 25 + (dy + 2) * 5 + (dx + 2)] * F(y + dy, x + dx);
  for (int dy = -1; dy <= 1; ++dy)
    for (int dx = -1; dx <= 1; ++dx)
      s += k3[c * 9 + (dy + 1) * 3 + (dx + 1)] * F(y + dy, x + dx);
  So[((long long)b * kSEQ + (kPAD + 1) + y * kPS + x) * kEMB + c] = s;
}

__global__ __launch_bounds__(256) void final_head_k(
    const float* __restrict__ S, const float* __restrict__ g, const float* __restrict__ bb,
    const float* __restrict__ w, const float* __restrict__ cb, float* __restrict__ out) {
  int b = blockIdx.x;
  const float* x = S + ((long long)b * kSEQ + kPAD) * kEMB;
  __shared__ float red[256];
  int t = threadIdx.x;
  float s1 = 0.f, s2 = 0.f;
  for (int c = t; c < kEMB; c += 256) { float v = x[c]; s1 += v; s2 += v * v; }
  red[t] = s1; __syncthreads();
  for (int s = 128; s > 0; s >>= 1) { if (t < s) red[t] += red[t + s]; __syncthreads(); }
  float mu = red[0] * (1.f / kEMB); __syncthreads();
  red[t] = s2; __syncthreads();
  for (int s = 128; s > 0; s >>= 1) { if (t < s) red[t] += red[t + s]; __syncthreads(); }
  float var = red[0] * (1.f / kEMB) - mu * mu; __syncthreads();
  float inv = rsqrtf(var + 1e-5f);
  float dot = 0.f;
  for (int c = t; c < kEMB; c += 256)
    dot += (((x[c] - mu) * inv) * g[c] + bb[c]) * w[c];
  red[t] = dot; __syncthreads();
  for (int s = 128; s > 0; s >>= 1) { if (t < s) red[t] += red[t + s]; __syncthreads(); }
  if (t == 0) out[b] = red[0] + cb[0];
}

// ---------------- host side ----------------
static inline void launch_gemm(hipStream_t st, const half_t* A, const half_t* Bp,
                               const float* bias, float* C, int M, int N, int K,
                               int lda, int ldb, int ldc,
                               long long sA, long long sB, long long sC,
                               int batch, bool transb) {
  dim3 grid((N + BNT - 1) / BNT, (M + BMT - 1) / BMT, batch);
  dim3 blk(256);
  if (transb) {
    if (bias) wmma_gemm<true, true><<<grid, blk, 0, st>>>(A, Bp, bias, C, M, N, K, lda, ldb, ldc, sA, sB, sC);
    else      wmma_gemm<true, false><<<grid, blk, 0, st>>>(A, Bp, bias, C, M, N, K, lda, ldb, ldc, sA, sB, sC);
  } else {
    if (bias) wmma_gemm<false, true><<<grid, blk, 0, st>>>(A, Bp, bias, C, M, N, K, lda, ldb, ldc, sA, sB, sC);
    else      wmma_gemm<false, false><<<grid, blk, 0, st>>>(A, Bp, bias, C, M, N, K, lda, ldb, ldc, sA, sB, sC);
  }
}

extern "C" void kernel_launch(void* const* d_in, const int* in_sizes, int n_in,
                              void* d_out, int out_size, void* d_ws, size_t ws_size,
                              hipStream_t stream) {
  (void)in_sizes; (void)n_in; (void)out_size; (void)ws_size;

  const float* X         = (const float*)d_in[0];
  const float* few_w     = (const float*)d_in[1];
  const float* few_b     = (const float*)d_in[2];
  const float* cls_token = (const float*)d_in[3];
  const float* ln_g[2]   = {(const float*)d_in[4],  (const float*)d_in[10]};
  const float* ln_b[2]   = {(const float*)d_in[5],  (const float*)d_in[11]};
  const float* qkv_w[2]  = {(const float*)d_in[6],  (const float*)d_in[12]};
  const float* out_w[2]  = {(const float*)d_in[7],  (const float*)d_in[13]};
  const float* out_bias[2] = {(const float*)d_in[8], (const float*)d_in[14]};
  const float* res_k[2]  = {(const float*)d_in[9],  (const float*)d_in[15]};
  const float* pk7 = (const float*)d_in[16]; const float* pb7 = (const float*)d_in[17];
  const float* pk5 = (const float*)d_in[18]; const float* pb5 = (const float*)d_in[19];
  const float* pk3 = (const float*)d_in[20]; const float* pb3 = (const float*)d_in[21];
  const float* norm_g = (const float*)d_in[22];
  const float* norm_b = (const float*)d_in[23];
  const float* cls_w  = (const float*)d_in[24];
  const float* cls_b  = (const float*)d_in[25];
  float* out = (float*)d_out;

  // ---- workspace arena ----
  char* wp = (char*)d_ws;
  auto alloc = [&](size_t bytes) -> char* {
    char* p = wp;
    wp += (bytes + 255) & ~(size_t)255;
    return p;
  };
  const long long SE = (long long)kB * kSEQ * kEMB;          // 17.8M
  float*  BUF0  = (float*)alloc(SE * 4);
  float*  BUF1  = (float*)alloc(SE * 4);
  float*  PROJ  = (float*)alloc(SE * 4);
  half_t* LNh   = (half_t*)alloc(SE * 2);
  half_t* Xh    = (half_t*)alloc((long long)kB * kBAG * kIN * 2);
  half_t* fewWh = (half_t*)alloc((long long)kIN * kEMB * 2);
  half_t* qkvWh = (half_t*)alloc((long long)kEMB * kQKV * 2);
  half_t* outWh = (half_t*)alloc((long long)kEMB * kEMB * 2);
  float*  QKVf  = (float*)alloc((long long)kB * kSEQ * kQKV * 4);
  float*  ATTf  = QKVf;  // alias: attention logits reuse qkv f32 buffer
  half_t* qh    = (half_t*)alloc((long long)kBH * kSEQ * kDH * 2);
  half_t* kh    = (half_t*)alloc((long long)kBH * kSEQ * kDH * 2);
  half_t* vh    = (half_t*)alloc((long long)kBH * kSEQ * kDH * 2);
  half_t* qlh   = (half_t*)alloc((long long)kBH * kNL * kDH * 2);
  half_t* klh   = (half_t*)alloc((long long)kBH * kNL * kDH * 2);
  half_t* a1h   = (half_t*)alloc((long long)kBH * kSEQ * kNL * 2);
  half_t* a3h   = (half_t*)alloc((long long)kBH * kSEQ * kNL * 2);
  float*  a2f   = (float*)alloc((long long)kBH * kNL * kNL * 4);
  half_t* a2h   = (half_t*)alloc((long long)kBH * kNL * kNL * 2);
  half_t* zh    = (half_t*)alloc((long long)kBH * kNL * kNL * 2);
  half_t* th    = (half_t*)alloc((long long)kBH * kNL * kNL * 2);
  half_t* xzh   = (half_t*)alloc((long long)kBH * kNL * kNL * 2);
  float*  P0f   = (float*)alloc((long long)kBH * kNL * kNL * 4);
  float*  P1f   = (float*)alloc((long long)kBH * kNL * kNL * 4);
  float*  P2f   = (float*)alloc((long long)kBH * kNL * kNL * 4);
  float*  a3vf  = (float*)alloc((long long)kBH * kNL * kDH * 4);
  half_t* a3vh  = (half_t*)alloc((long long)kBH * kNL * kDH * 2);
  float*  pzf   = (float*)alloc((long long)kBH * kNL * kDH * 4);
  half_t* pzh   = (half_t*)alloc((long long)kBH * kNL * kDH * 2);
  float*  ATTNf = (float*)alloc((long long)kBH * kSEQ * kDH * 4);
  half_t* Mh    = (half_t*)alloc(SE * 2);

  const float scale = 1.0f / sqrtf((float)kDH);
  auto blocks1d = [](long long n) { return (unsigned)((n + 255) / 256); };
  auto cvt = [&](const float* src, half_t* dst, long long n, float s) {
    unsigned g = blocks1d(n); if (g > 65535u * 8u) g = 65535u * 8u;
    cvt_scale_k<<<g, 256, 0, stream>>>(src, dst, s, n);
  };

  // ---- embedding: h = X @ few_w + few_b, ps=64 so add=0; cls at row kPAD ----
  cvt(X, Xh, (long long)kB * kBAG * kIN, 1.f);
  cvt(few_w, fewWh, (long long)kIN * kEMB, 1.f);
  launch_gemm(stream, Xh, fewWh, few_b, BUF0 + (long long)(kPAD + 1) * kEMB,
              kBAG, kEMB, kIN, kIN, kEMB, kEMB,
              (long long)kBAG * kIN, 0, (long long)kSEQ * kEMB, kB, false);
  set_cls_k<<<kB, 256, 0, stream>>>(BUF0, cls_token);

  // ---- one Nystrom attention layer ----
  auto run_layer = [&](float* S, int li) {
    cvt(qkv_w[li], qkvWh, (long long)kEMB * kQKV, 1.f);
    cvt(out_w[li], outWh, (long long)kEMB * kEMB, 1.f);

    layernorm_to_h_k<<<kB * kSEQ, 256, 0, stream>>>(S, ln_g[li], ln_b[li], LNh);

    // qkv = LN(h) @ W_qkv   [34816,512]x[512,1536]
    launch_gemm(stream, LNh, qkvWh, nullptr, QKVf,
                kB * kSEQ, kQKV, kEMB, kEMB, kQKV, kQKV, 0, 0, 0, 1, false);
    split_heads_k<<<blocks1d(SE), 256, 0, stream>>>(QKVf, qh, kh, vh, scale);
    landmark_mean_k<<<blocks1d((long long)kBH * kNL * kDH), 256, 0, stream>>>(qh, qlh);
    landmark_mean_k<<<blocks1d((long long)kBH * kNL * kDH), 256, 0, stream>>>(kh, klh);

    // a1 = softmax(q @ k_l^T)   batch=32: [4352,128]x[256,128]^T
    launch_gemm(stream, qh, klh, nullptr, ATTf, kSEQ, kNL, kDH, kDH, kDH, kNL,
                (long long)kSEQ * kDH, (long long)kNL * kDH, (long long)kSEQ * kNL, kBH, true);
    softmax_rows_k<<<kBH * kSEQ, 256, 0, stream>>>(ATTf, a1h, kNL);

    // a2 = softmax(q_l @ k_l^T)
    launch_gemm(stream, qlh, klh, nullptr, a2f, kNL, kNL, kDH, kDH, kDH, kNL,
                (long long)kNL * kDH, (long long)kNL * kDH, (long long)kNL * kNL, kBH, true);
    softmax_rows_k<<<kBH * kNL, 256, 0, stream>>>(a2f, a2h, kNL);

    // pinv init + 6 Newton-Schulz iterations (batched 256^3 WMMA GEMMs)
    pinv_init_k<<<kBH, 256, 0, stream>>>(a2f, zh);
    const long long pin = (long long)kBH * kNL * kNL;
    for (int it = 0; it < 6; ++it) {
      launch_gemm(stream, a2h, zh, nullptr, P0f, kNL, kNL, kNL, kNL, kNL, kNL,
                  pin, pin, pin, kBH, false);                         // xz
      cvt(P0f, xzh, pin, 1.f);
      aI_minus_k<<<blocks1d(pin), 256, 0, stream>>>(P0f, th, 7.f, pin);
      launch_gemm(stream, xzh, th, nullptr, P1f, kNL, kNL, kNL, kNL, kNL, kNL,
                  pin, pin, pin, kBH, false);                         // xz(7I-xz)
      aI_minus_k<<<blocks1d(pin), 256, 0, stream>>>(P1f, th, 15.f, pin);
      launch_gemm(stream, xzh, th, nullptr, P2f, kNL, kNL, kNL, kNL, kNL, kNL,
                  pin, pin, pin, kBH, false);                         // xz(15I-...)
      aI_minus_k<<<blocks1d(pin), 256, 0, stream>>>(P2f, th, 13.f, pin);
      launch_gemm(stream, zh, th, nullptr, P0f, kNL, kNL, kNL, kNL, kNL, kNL,
                  pin, pin, pin, kBH, false);                         // z(13I-...)
      cvt(P0f, zh, pin, 0.25f);
    }

    // a3 = softmax(q_l @ k^T)   [256,128]x[4352,128]^T
    launch_gemm(stream, qlh, kh, nullptr, ATTf, kNL, kSEQ, kDH, kDH, kDH, kSEQ,
                (long long)kNL * kDH, (long long)kSEQ * kDH, (long long)kNL * kSEQ, kBH, true);
    softmax_rows_k<<<kBH * kNL, 256, 0, stream>>>(ATTf, a3h, kSEQ);

    // a3v = a3 @ v   [256,4352]x[4352,128]
    launch_gemm(stream, a3h, vh, nullptr, a3vf, kNL, kDH, kSEQ, kSEQ, kDH, kDH,
                (long long)kNL * kSEQ, (long long)kSEQ * kDH, (long long)kNL * kDH, kBH, false);
    cvt(a3vf, a3vh, (long long)kBH * kNL * kDH, 1.f);

    // pz = pinv(a2) @ a3v   [256,256]x[256,128]
    launch_gemm(stream, zh, a3vh, nullptr, pzf, kNL, kDH, kNL, kNL, kDH, kDH,
                pin, (long long)kNL * kDH, (long long)kNL * kDH, kBH, false);
    cvt(pzf, pzh, (long long)kBH * kNL * kDH, 1.f);

    // attn = a1 @ pz   [4352,256]x[256,128]
    launch_gemm(stream, a1h, pzh, nullptr, ATTNf, kSEQ, kDH, kNL, kNL, kDH, kDH,
                (long long)kSEQ * kNL, (long long)kNL * kDH, (long long)kSEQ * kDH, kBH, false);

    // + 33-tap depthwise residual conv on v, merge heads -> Mh (f16)
    merge_res_k<<<blocks1d(SE), 256, 0, stream>>>(ATTNf, vh, res_k[li], Mh);

    // out-proj + bias, then residual add into S (rows >= kPAD only)
    launch_gemm(stream, Mh, outWh, out_bias[li], PROJ,
                kB * kSEQ, kEMB, kEMB, kEMB, kEMB, kEMB, 0, 0, 0, 1, false);
    add_residual_k<<<blocks1d(SE), 256, 0, stream>>>(S, PROJ);
  };

  run_layer(BUF0, 0);

  // ---- PPEG: f + dw7 + dw5 + dw3 on 64x64 grid; cls row passes through ----
  ppeg_k<<<blocks1d((long long)kB * kEMB * kPS * kPS), 256, 0, stream>>>(
      BUF0, BUF1, pk7, pb7, pk5, pb5, pk3, pb3);
  copy_cls_k<<<kB, 256, 0, stream>>>(BUF0, BUF1);

  run_layer(BUF1, 1);

  // ---- final LN + classifier on cls token ----
  final_head_k<<<kB, 256, 0, stream>>>(BUF1, norm_g, norm_b, cls_w, cls_b, out);
}
